// Attention_Scatter_51196010168975
// MI455X (gfx1250) — matrix operand order repaired and checked by reading, not compiled, not run on hardware
//
#include <hip/hip_runtime.h>
#include <hip/hip_bf16.h>

typedef __attribute__((ext_vector_type(16))) _Float16 v16h;
typedef __attribute__((ext_vector_type(8)))  float    v8f;
typedef __attribute__((ext_vector_type(4)))  float    v4f;

#define D 512
#define NTILES 32     // D/16 column tiles
#define KSTEPS 16     // D/32 k-steps
#define NFRAGS (NTILES * KSTEPS)   // 512
#define BBUF_DEPTH 4               // B-stream rotation depth (power of 2)
#define FRAG_PAD BBUF_DEPTH        // extra fragments so prefetch is branch-free
#define GEMM_WAVES 8

#if defined(__has_builtin) && __has_builtin(__builtin_amdgcn_sched_group_barrier)
#define SCHED_GROUP(mask, size) __builtin_amdgcn_sched_group_barrier((mask), (size), 0)
#else
#define SCHED_GROUP(mask, size)
#endif
// mask bits: 0x020 = VMEM read, 0x008 = MFMA/WMMA

// ---------------------------------------------------------------------------
// Pack W (f32, [K=512, N=512] row-major) into f16 WMMA B-fragments, and pack
// (b,u) into a float2 array with 16 entries of zero padding.
// Fragment (n,k) covers K in [32k,32k+32), N in [16n,16n+16).
// B layout (16-bit, 32x16): lanes 0-15:  N=lane,     h<8 -> K = 32k + h, else K = 32k + 8 + h
//                           lanes 16-31: N=lane-16,  h<8 -> K = 32k + 8 + h, else K = 32k + 16 + h
// Stored at wpack[((n*16+k)*32 + lane)*16 + h].
// ---------------------------------------------------------------------------
__global__ void pack_kernel(const float* __restrict__ w,
                            const float* __restrict__ b,
                            const float* __restrict__ u,
                            _Float16* __restrict__ wpack,
                            float2* __restrict__ bu) {
    int t = blockIdx.x * blockDim.x + threadIdx.x;
    if (t < NFRAGS * 32) {
        int frag = t >> 5;          // n*16 + k
        int lane = t & 31;
        int n = frag >> 4;
        int k = frag & 15;
        int col   = n * 16 + (lane & 15);
        int kbase = 32 * k + ((lane >> 4) ? 8 : 0);
        _Float16* dst = wpack + ((size_t)frag * 32 + lane) * 16;
#pragma unroll
        for (int h = 0; h < 16; ++h)
            dst[h] = (_Float16)w[(size_t)(kbase + h + (h >= 8 ? 8 : 0)) * D + col];
    } else {
        int i = t - NFRAGS * 32;
        if (i < D + 16)
            bu[i] = (i < D) ? make_float2(b[i], u[i]) : make_float2(0.f, 0.f);
    }
}

// ---------------------------------------------------------------------------
// Fused GEMM + tanh + u-projection. One wave per 16-row tile of x.
// vu[i] = sum_c tanh( (x@W)[i,c] + b[c] ) * u[c]
// A tile lives in 16 register-resident v16h fragments; B fragments are
// software-pipelined BBUF_DEPTH deep, with sched_group_barrier enforcing a
// 2xVMEM-read / 1xWMMA interleave so the scheduler cannot sink the prefetch.
// ---------------------------------------------------------------------------
__global__ __launch_bounds__(GEMM_WAVES * 32)
void gemm_tanh_vu_kernel(const float* __restrict__ x,
                         const _Float16* __restrict__ wpack,
                         const float2* __restrict__ bu,
                         float* __restrict__ vu, int nRowTiles) {
    const int wave = threadIdx.x >> 5;
    const int lane = threadIdx.x & 31;
    const int tile = blockIdx.x * GEMM_WAVES + wave;
    if (tile >= nRowTiles) return;   // wave-uniform: EXEC stays all-1 for live waves

    // --- Build A fragments in registers.
    // A layout (16-bit, 16x32): lanes 0-15:  M=lane,    K = {0..7, 16..23}
    //                           lanes 16-31: M=lane-16, K = {8..15, 24..31}
    v16h afr[KSTEPS];
    {
        const int r = lane & 15;
        const int hsel = (lane >> 4) ? 8 : 0;
        const float* xrow = x + (size_t)(tile * 16 + r) * D;
#pragma unroll
        for (int k = 0; k < KSTEPS; ++k) {
            const int c0 = 32 * k + hsel;
            v4f f0 = *(const v4f*)(xrow + c0);
            v4f f1 = *(const v4f*)(xrow + c0 + 4);
            v4f f2 = *(const v4f*)(xrow + c0 + 16);
            v4f f3 = *(const v4f*)(xrow + c0 + 20);
#pragma unroll
            for (int j = 0; j < 4; ++j) {
                afr[k][j]      = (_Float16)f0[j];
                afr[k][4 + j]  = (_Float16)f1[j];
                afr[k][8 + j]  = (_Float16)f2[j];
                afr[k][12 + j] = (_Float16)f3[j];
            }
        }
    }

    // Per-lane base of the B-fragment stream; fragment f at +f*512 halves.
    const _Float16* bbase = wpack + (size_t)lane * 16;
#define LOAD_FRAG(f) (*(const v16h*)(bbase + (size_t)(f) * (32 * 16)))

    // --- Software-pipelined B stream: BBUF_DEPTH fragments in flight.
    v16h bbuf[BBUF_DEPTH];
#pragma unroll
    for (int f = 0; f < BBUF_DEPTH; ++f) bbuf[f] = LOAD_FRAG(f);

    float rowpart[8];
#pragma unroll
    for (int v = 0; v < 8; ++v) rowpart[v] = 0.0f;

    const int cLane = lane & 15;
    float2 bu_cur = bu[cLane];                 // (b,u) for n = 0

    for (int n = 0; n < NTILES; ++n) {
        float2 bu_next = bu[(n + 1) * 16 + cLane];   // padded: always in bounds
        SCHED_GROUP(0x020, 1);                       // keep the bu load up front
        v8f acc = {};
#pragma unroll
        for (int k = 0; k < KSTEPS; ++k) {
            const int f = n * KSTEPS + k;
            v16h cur = bbuf[f & (BBUF_DEPTH - 1)];
            bbuf[f & (BBUF_DEPTH - 1)] = LOAD_FRAG(f + BBUF_DEPTH); // padded
            acc = __builtin_amdgcn_wmma_f32_16x16x32_f16(
                false, afr[k], false, cur, (short)0, acc, false, false);
            SCHED_GROUP(0x020, 2);   // 2 VMEM reads (prefetch of f+4)
            SCHED_GROUP(0x008, 1);   // 1 WMMA (consumes f)
        }
        // C/D layout: lanes 0-15: (M=v, N=lane); lanes 16-31: (M=8+v, N=lane-16)
        const float bc = bu_cur.x;
        const float uc = bu_cur.y;
#pragma unroll
        for (int v = 0; v < 8; ++v) {
            float t  = __expf(2.0f * (acc[v] + bc));
            float th = (t - 1.0f) / (t + 1.0f);   // tanh
            rowpart[v] += th * uc;
        }
        bu_cur = bu_next;
    }
#undef LOAD_FRAG

    // --- Reduce across the 16 lanes of each half-wave (columns of the tile).
#pragma unroll
    for (int v = 0; v < 8; ++v) {
        float s = rowpart[v];
        s += __shfl_xor(s, 1, 32);
        s += __shfl_xor(s, 2, 32);
        s += __shfl_xor(s, 4, 32);
        s += __shfl_xor(s, 8, 32);
        rowpart[v] = s;
    }
    if (lane == 0) {
#pragma unroll
        for (int v = 0; v < 8; ++v) vu[tile * 16 + v] = rowpart[v];
    }
    if (lane == 16) {
#pragma unroll
        for (int v = 0; v < 8; ++v) vu[tile * 16 + 8 + v] = rowpart[v];
    }
}

// ---------------------------------------------------------------------------
// Segment boundaries via lower_bound (ids are sorted). start[S] == N.
// ---------------------------------------------------------------------------
__global__ void seg_bounds_kernel(const int* __restrict__ idx, int* __restrict__ start,
                                  int n, int S) {
    int s = blockIdx.x * blockDim.x + threadIdx.x;
    if (s > S) return;
    int lo = 0, hi = n;
    while (lo < hi) {
        int mid = (lo + hi) >> 1;
        if (idx[mid] < s) lo = mid + 1; else hi = mid;
    }
    start[s] = lo;
}

// ---------------------------------------------------------------------------
// One block per segment: stable softmax over vu rows, write alphas, and
// register-accumulated weighted pooling out[s,:] = sum x[i,:]*alpha_i.
// ---------------------------------------------------------------------------
__global__ __launch_bounds__(256)
void seg_softmax_pool_kernel(const float* __restrict__ x,
                             const float* __restrict__ vu,
                             const int* __restrict__ start,
                             float* __restrict__ out,
                             float* __restrict__ alphas) {
    __shared__ float red[256];
    __shared__ float wbuf[256];
    const int s = blockIdx.x;
    const int t = threadIdx.x;
    const int a = start[s];
    const int cnt = start[s + 1] - a;

    // max
    float m = -3.402823466e38f;
    for (int j = t; j < cnt; j += 256) m = fmaxf(m, vu[a + j]);
    red[t] = m; __syncthreads();
    for (int ofs = 128; ofs > 0; ofs >>= 1) {
        if (t < ofs) red[t] = fmaxf(red[t], red[t + ofs]);
        __syncthreads();
    }
    m = red[0]; __syncthreads();

    // sum of exp
    float ssum = 0.0f;
    for (int j = t; j < cnt; j += 256) ssum += __expf(vu[a + j] - m);
    red[t] = ssum; __syncthreads();
    for (int ofs = 128; ofs > 0; ofs >>= 1) {
        if (t < ofs) red[t] += red[t + ofs];
        __syncthreads();
    }
    ssum = red[0]; __syncthreads();
    const float inv = (cnt > 0) ? 1.0f / ssum : 0.0f;

    // pooling: each thread owns 2 columns
    float acc0 = 0.0f, acc1 = 0.0f;
    const int c0 = t, c1 = t + 256;
    for (int base = 0; base < cnt; base += 256) {
        const int lim = min(256, cnt - base);
        if (t < lim) {
            float w = __expf(vu[a + base + t] - m) * inv;
            wbuf[t] = w;
            alphas[a + base + t] = w;
        }
        __syncthreads();
        for (int j = 0; j < lim; ++j) {
            const float w = wbuf[j];
            const float* xr = x + (size_t)(a + base + j) * D;
            acc0 += xr[c0] * w;
            acc1 += xr[c1] * w;
        }
        __syncthreads();
    }
    out[(size_t)s * D + c0] = acc0;
    out[(size_t)s * D + c1] = acc1;
}

// ---------------------------------------------------------------------------
extern "C" void kernel_launch(void* const* d_in, const int* in_sizes, int n_in,
                              void* d_out, int out_size, void* d_ws, size_t ws_size,
                              hipStream_t stream) {
    const float* x   = (const float*)d_in[0];
    const float* w   = (const float*)d_in[1];
    const float* b   = (const float*)d_in[2];
    const float* u   = (const float*)d_in[3];
    const int*   idx = (const int*)d_in[4];

    const int N = in_sizes[0] / D;          // 200000
    const int S = (out_size - N) / D;       // 10000 (out = [S*D] ++ [N])

    // workspace layout
    char* wsb = (char*)d_ws;
    _Float16* wpack = (_Float16*)wsb;       // (512+pad) fragments * 1KB
    size_t off = (size_t)(NFRAGS + FRAG_PAD) * 32 * 16 * sizeof(_Float16);
    float2* bu = (float2*)(wsb + off);      // (512+16) float2
    off += (size_t)(D + 16) * sizeof(float2);
    off = (off + 255) & ~(size_t)255;
    float* vu = (float*)(wsb + off);        // N*4
    off += ((size_t)N * sizeof(float) + 255) & ~(size_t)255;
    int* segstart = (int*)(wsb + off);      // (S+1)*4

    float* out    = (float*)d_out;
    float* alphas = (float*)d_out + (size_t)S * D;

    // 1) pack W into f16 B-fragments + (b,u) pairs
    {
        int threads = NFRAGS * 32 + (D + 16);   // 16384 + 528
        pack_kernel<<<(threads + 255) / 256, 256, 0, stream>>>(w, b, u, wpack, bu);
    }
    // 2) segment boundaries (independent of 1)
    seg_bounds_kernel<<<(S + 1 + 255) / 256, 256, 0, stream>>>(idx, segstart, N, S);
    // 3) fused WMMA GEMM + tanh + u-projection -> vu
    {
        int nRowTiles = N / 16;               // 12500 (N divisible by 16)
        int blocks = (nRowTiles + GEMM_WAVES - 1) / GEMM_WAVES;
        gemm_tanh_vu_kernel<<<blocks, GEMM_WAVES * 32, 0, stream>>>(
            x, wpack, bu, vu, nRowTiles);
    }
    // 4) segment softmax + weighted pooling
    seg_softmax_pool_kernel<<<S, 256, 0, stream>>>(x, vu, segstart, out, alphas);
}
